// RNN_2267742732750
// MI455X (gfx1250) — matrix operand order repaired and checked by reading, not compiled, not run on hardware
//
#include <hip/hip_runtime.h>
#include <hip/hip_bf16.h>

// ---------------------------------------------------------------------------
// Fused LSTM cell for MI455X (gfx1250):
//   gates = [x | h_old] (16384x1536) @ [U;W] (1536x4096) + b
//   i,f,o = sigmoid ; g = tanh ; c = i*g + f*c_old ; h = o*tanh(c)
//
// bf16 pre-pack (fits in 192MB L2) + v_wmma_f32_16x16x32_bf16 GEMM with fused
// LSTM epilogue. Block tile 256x128 (N = 4 gates x 32 h-cols, so each wave
// owns all four gates for its rows). Wave tile 32x128 = 16 WMMA tiles ->
// 1.25 ds_load_b128 per WMMA. Double-buffered LDS: one barrier per k-step.
// ---------------------------------------------------------------------------

typedef __attribute__((ext_vector_type(16))) __bf16 v16bf;
typedef __attribute__((ext_vector_type(8)))  float  v8f;

#define B_ROWS 16384
#define D_IN   512
#define H_DIM  1024
#define K_TOT  1536          // D_IN + H
#define N_TOT  4096          // 4 * H

#define BM 256               // rows per block
#define BN 128               // 4 gates x 32 h-cols
#define BK 32
#define AROW 40              // padded LDS row stride in bf16 elements (80B)
#define NSTEP (K_TOT / BK)   // 48

union Frag {
  v16bf v;
  uint4 q[2];
};

__device__ __forceinline__ unsigned short f2bf(float f) {
  unsigned u = __float_as_uint(f);
  u += 0x7FFFu + ((u >> 16) & 1u);    // round-to-nearest-even bf16
  return (unsigned short)(u >> 16);
}

__device__ __forceinline__ float sigm_(float x) {
  return 1.0f / (1.0f + __expf(-x));
}
__device__ __forceinline__ float tanh_(float x) {
  return 1.0f - 2.0f / (1.0f + __expf(2.0f * x));
}

// ---- prep: pack [x | h_old] -> bf16 row-major [16384][1536] ---------------
__global__ void pack_a_kernel(const float* __restrict__ x,
                              const float* __restrict__ h,
                              unsigned short* __restrict__ Abf) {
  int t  = blockIdx.x * blockDim.x + threadIdx.x;   // 16384 * 192 threads
  int b  = t / (K_TOT / 8);
  int kc = (t % (K_TOT / 8)) * 8;
  const float* src = (kc < D_IN) ? (x + (size_t)b * D_IN + kc)
                                 : (h + (size_t)b * H_DIM + (kc - D_IN));
  float4 f0 = ((const float4*)src)[0];
  float4 f1 = ((const float4*)src)[1];
  alignas(16) unsigned short r[8];
  r[0] = f2bf(f0.x); r[1] = f2bf(f0.y); r[2] = f2bf(f0.z); r[3] = f2bf(f0.w);
  r[4] = f2bf(f1.x); r[5] = f2bf(f1.y); r[6] = f2bf(f1.z); r[7] = f2bf(f1.w);
  *(uint4*)(Abf + (size_t)b * K_TOT + kc) = *(const uint4*)r;
}

// ---- prep: transpose-pack [U;W] -> bf16 Bt[n][k], n = gate*1024 + h -------
__global__ void pack_b_kernel(const float* __restrict__ U0, const float* __restrict__ U1,
                              const float* __restrict__ U2, const float* __restrict__ U3,
                              const float* __restrict__ W0, const float* __restrict__ W1,
                              const float* __restrict__ W2, const float* __restrict__ W3,
                              unsigned short* __restrict__ Bt) {
  __shared__ unsigned short tile[32][33];
  int k0   = blockIdx.x * 32;
  int n0   = blockIdx.y * 32;
  int gate = n0 >> 10;
  int hh   = n0 & (H_DIM - 1);
  const float* Us[4] = {U0, U1, U2, U3};
  const float* Ws[4] = {W0, W1, W2, W3};
  const float* src = (k0 < D_IN) ? (Us[gate] + (size_t)k0 * H_DIM + hh)
                                 : (Ws[gate] + (size_t)(k0 - D_IN) * H_DIM + hh);
  int tx = threadIdx.x, ty = threadIdx.y;
#pragma unroll
  for (int j = 0; j < 4; ++j) {
    int kk = ty + 8 * j;
    tile[kk][tx] = f2bf(src[(size_t)kk * H_DIM + tx]);   // coalesced read
  }
  __syncthreads();
#pragma unroll
  for (int j = 0; j < 4; ++j) {
    int nn = ty + 8 * j;
    Bt[(size_t)(n0 + nn) * K_TOT + k0 + tx] = tile[tx][nn];  // coalesced write
  }
}

// ---- prep: bias[gate*1024 + h] = U*_b + W*_b ------------------------------
__global__ void pack_bias_kernel(const float* __restrict__ u0, const float* __restrict__ u1,
                                 const float* __restrict__ u2, const float* __restrict__ u3,
                                 const float* __restrict__ w0, const float* __restrict__ w1,
                                 const float* __restrict__ w2, const float* __restrict__ w3,
                                 float* __restrict__ biasp) {
  int t = blockIdx.x * blockDim.x + threadIdx.x;   // 0..4095
  int g = t >> 10, hh = t & (H_DIM - 1);
  const float* bu[4] = {u0, u1, u2, u3};
  const float* bw[4] = {w0, w1, w2, w3};
  biasp[t] = bu[g][hh] + bw[g][hh];
}

// ---- main: bf16 WMMA GEMM + fused LSTM epilogue ---------------------------
__global__ __launch_bounds__(256)
void lstm_gemm_kernel(const unsigned short* __restrict__ Abf,
                      const unsigned short* __restrict__ Bt,
                      const float* __restrict__ biasp,
                      const float* __restrict__ c_old,
                      float* __restrict__ h_out,
                      float* __restrict__ c_out) {
  // double-buffered stage: A 2x20KB, B 2x10KB = 60KB (of 320KB/WGP)
  __shared__ __align__(16) unsigned short Alds[2][BM * AROW];
  __shared__ __align__(16) unsigned short Blds[2][BN * AROW];

  const int tid = threadIdx.x;
  const int m0  = blockIdx.x * BM;   // batch rows
  const int h0  = blockIdx.y * 32;   // h-columns (per gate)

  // global->LDS copy mapping:
  //   A: 1 thread per row, 32 bf16 (64B) each
  //   B: 2 threads per row, 16 bf16 (32B) each
  const int brow  = tid >> 1;         // 0..127
  const int bhalf = (tid & 1) * 16;

  const unsigned short* Ag = Abf + (size_t)(m0 + tid) * K_TOT;
  const int gate_c = brow >> 5;
  const int hc_c   = brow & 31;
  const unsigned short* Bg =
      Bt + (size_t)(gate_c * H_DIM + h0 + hc_c) * K_TOT + bhalf;

  const int aoff = tid * AROW;                 // A LDS store base (elements)
  const int boff = brow * AROW + bhalf;        // B LDS store base

  // per-wave fragment mapping (wave32): wave wv owns rows 32*wv..32*wv+31
  const int lane = tid & 31;
  const int wv   = tid >> 5;
  const int lm   = lane & 15;
  const int lh   = lane >> 4;         // K-half selector per ISA layout

  v8f acc[2][8];
#pragma unroll
  for (int j = 0; j < 2; ++j)
#pragma unroll
    for (int t = 0; t < 8; ++t)
#pragma unroll
      for (int e = 0; e < 8; ++e) acc[j][t][e] = 0.0f;

  // ---- prologue: stage k-step 0 ----
  uint4 ar0 = ((const uint4*)Ag)[0];
  uint4 ar1 = ((const uint4*)Ag)[1];
  uint4 ar2 = ((const uint4*)Ag)[2];
  uint4 ar3 = ((const uint4*)Ag)[3];
  uint4 br0 = ((const uint4*)Bg)[0];
  uint4 br1 = ((const uint4*)Bg)[1];
  {
    uint4* as = (uint4*)&Alds[0][aoff];
    as[0] = ar0; as[1] = ar1; as[2] = ar2; as[3] = ar3;
    uint4* bs = (uint4*)&Blds[0][boff];
    bs[0] = br0; bs[1] = br1;
  }
  __syncthreads();

  for (int ks = 0; ks < NSTEP; ++ks) {
    const int cur  = ks & 1;
    const int nxt  = cur ^ 1;
    const bool more = (ks + 1) < NSTEP;

    if (more) {
      const unsigned short* An = Ag + (ks + 1) * BK;
      const unsigned short* Bn = Bg + (ks + 1) * BK;
      ar0 = ((const uint4*)An)[0]; ar1 = ((const uint4*)An)[1];
      ar2 = ((const uint4*)An)[2]; ar3 = ((const uint4*)An)[3];
      br0 = ((const uint4*)Bn)[0]; br1 = ((const uint4*)Bn)[1];
      if (ks + 2 < NSTEP) {                       // -> global_prefetch_b8
        __builtin_prefetch(Ag + (ks + 2) * BK, 0, 1);
        __builtin_prefetch(Bg + (ks + 2) * BK, 0, 1);
      }
    }

    // A fragments 16x32 bf16: lane lm = row; chunks at K=8*lh and K=16+8*lh
    Frag af0, af1;
    {
      const int r0 = (32 * wv + lm) * AROW;
      const int r1 = (32 * wv + 16 + lm) * AROW;
      af0.q[0] = *(const uint4*)&Alds[cur][r0 + 8 * lh];
      af0.q[1] = *(const uint4*)&Alds[cur][r0 + 16 + 8 * lh];
      af1.q[0] = *(const uint4*)&Alds[cur][r1 + 8 * lh];
      af1.q[1] = *(const uint4*)&Alds[cur][r1 + 16 + 8 * lh];
    }

#pragma unroll
    for (int t = 0; t < 8; ++t) {
      // B fragment 32x16 bf16: lane lm = col; chunks at K=8*lh and K=16+8*lh
      Frag bfv;
      const int nb = (16 * t + lm) * AROW + 8 * lh;
      bfv.q[0] = *(const uint4*)&Blds[cur][nb];
      bfv.q[1] = *(const uint4*)&Blds[cur][nb + 16];
      acc[0][t] = __builtin_amdgcn_wmma_f32_16x16x32_bf16(
          false, af0.v, false, bfv.v, (short)0, acc[0][t], false, false);
      acc[1][t] = __builtin_amdgcn_wmma_f32_16x16x32_bf16(
          false, af1.v, false, bfv.v, (short)0, acc[1][t], false, false);
    }

    if (more) {
      uint4* as = (uint4*)&Alds[nxt][aoff];
      as[0] = ar0; as[1] = ar1; as[2] = ar2; as[3] = ar3;
      uint4* bs = (uint4*)&Blds[nxt][boff];
      bs[0] = br0; bs[1] = br1;
    }
    __syncthreads();
  }

  // ---- fused LSTM epilogue ----
  // tile (j, t=2*gate+u); element (v, lane):
  //   row = 32*wv + 16*j + v + 8*lh ; col-in-gate = 16*u + lm
#pragma unroll
  for (int u = 0; u < 2; ++u) {
    const int col = h0 + 16 * u + lm;
    const float bi = biasp[col];
    const float bf = biasp[H_DIM + col];
    const float bo = biasp[2 * H_DIM + col];
    const float bg = biasp[3 * H_DIM + col];
#pragma unroll
    for (int j = 0; j < 2; ++j) {
#pragma unroll
      for (int v = 0; v < 8; ++v) {
        const int r = m0 + 32 * wv + 16 * j + v + 8 * lh;
        const size_t idx = (size_t)r * H_DIM + col;
        const float co = c_old[idx];
        const float ig = sigm_(acc[j][0 + u][v] + bi);
        const float fg = sigm_(acc[j][2 + u][v] + bf);
        const float og = sigm_(acc[j][4 + u][v] + bo);
        const float gg = tanh_(acc[j][6 + u][v] + bg);
        const float cn = ig * gg + fg * co;
        c_out[idx] = cn;
        h_out[idx] = og * tanh_(cn);
      }
    }
  }
}

extern "C" void kernel_launch(void* const* d_in, const int* in_sizes, int n_in,
                              void* d_out, int out_size, void* d_ws, size_t ws_size,
                              hipStream_t stream) {
  const float* x    = (const float*)d_in[0];
  const float* hold = (const float*)d_in[1];
  const float* cold = (const float*)d_in[2];
  const float* Ui_w = (const float*)d_in[3];  const float* Ui_b = (const float*)d_in[4];
  const float* Uf_w = (const float*)d_in[5];  const float* Uf_b = (const float*)d_in[6];
  const float* Uo_w = (const float*)d_in[7];  const float* Uo_b = (const float*)d_in[8];
  const float* Uc_w = (const float*)d_in[9];  const float* Uc_b = (const float*)d_in[10];
  const float* Wi_w = (const float*)d_in[11]; const float* Wi_b = (const float*)d_in[12];
  const float* Wf_w = (const float*)d_in[13]; const float* Wf_b = (const float*)d_in[14];
  const float* Wo_w = (const float*)d_in[15]; const float* Wo_b = (const float*)d_in[16];
  const float* Wc_w = (const float*)d_in[17]; const float* Wc_b = (const float*)d_in[18];

  // workspace layout: Abf (48MB bf16) | Bt (12MB bf16) | biasp (16KB f32)
  unsigned short* Abf = (unsigned short*)d_ws;
  unsigned short* Bt  = Abf + (size_t)B_ROWS * K_TOT;
  float*          bp  = (float*)(Bt + (size_t)N_TOT * K_TOT);

  pack_a_kernel<<<(B_ROWS * (K_TOT / 8)) / 256, 256, 0, stream>>>(x, hold, Abf);
  pack_b_kernel<<<dim3(K_TOT / 32, N_TOT / 32), dim3(32, 8), 0, stream>>>(
      Ui_w, Uf_w, Uo_w, Uc_w, Wi_w, Wf_w, Wo_w, Wc_w, Bt);
  pack_bias_kernel<<<N_TOT / 256, 256, 0, stream>>>(
      Ui_b, Uf_b, Uo_b, Uc_b, Wi_b, Wf_b, Wo_b, Wc_b, bp);

  float* h_out = (float*)d_out;
  float* c_out = h_out + (size_t)B_ROWS * H_DIM;
  lstm_gemm_kernel<<<dim3(B_ROWS / BM, H_DIM / 32), 256, 0, stream>>>(
      Abf, Bt, bp, cold, h_out, c_out);
}
